// MultiHeadAttentionWithProbMask_31250182045977
// MI455X (gfx1250) — matrix operand order repaired and checked by reading, not compiled, not run on hardware
//
#include <hip/hip_runtime.h>
#include <hip/hip_bf16.h>
#include <stdint.h>

// ---------------------------------------------------------------------------
// MI455X (gfx1250) ProbSparse multi-head attention, bf16 WMMA + f32 accum.
// B=2, L=2048, D=512, H=8, dh=64, n_top = int(np.log(2048)) = 7.
// ---------------------------------------------------------------------------

typedef __bf16 bf16_t;
typedef __attribute__((ext_vector_type(16))) __bf16 v16bf;
typedef __attribute__((ext_vector_type(8)))  float  v8f;

#define DEVI __device__ __forceinline__

#define BATCH   2
#define LSEQ    2048
#define DMODEL  512
#define NHEAD   8
#define DHEAD   64
#define BH      (BATCH * NHEAD)     // 16
#define NTOP    7
#define ROWS    (BATCH * LSEQ)      // 4096 fused (b,l) rows

// ---- CDNA5 async global->LDS copy (ASYNCcnt path), guarded --------------
#ifndef __has_builtin
#define __has_builtin(x) 0
#endif
#if __has_builtin(__builtin_amdgcn_global_load_async_to_lds_b128)
#define HAVE_ASYNC 1
#else
#define HAVE_ASYNC 0
#endif

#if HAVE_ASYNC
// Signature (from toolchain diagnostic): (v4i AS1* src, v4i AS3* dst, Ii, Ii)
typedef int v4i_vs __attribute__((vector_size(16)));
typedef __attribute__((address_space(1))) v4i_vs as1_v4i;
typedef __attribute__((address_space(3))) v4i_vs as3_v4i;
DEVI void async_cp16(void* lds_dst, const void* gsrc) {
  __builtin_amdgcn_global_load_async_to_lds_b128(
      (as1_v4i*)gsrc, (as3_v4i*)lds_dst, /*imm_offset=*/0, /*cpol=*/0);
}
DEVI void async_wait_all() {
#if __has_builtin(__builtin_amdgcn_s_wait_asynccnt)
  __builtin_amdgcn_s_wait_asynccnt(0);
#else
  asm volatile("s_wait_asynccnt 0" ::: "memory");
#endif
}
#else
DEVI void async_cp16(void* lds_dst, const void* gsrc) {
  *(uint4*)lds_dst = *(const uint4*)gsrc;   // VGPR relay fallback
}
DEVI void async_wait_all() {}
#endif

DEVI bf16_t f2bf(float f) {
  union { float f; unsigned u; } v; v.f = f;
  unsigned r = v.u + 0x7FFFu + ((v.u >> 16) & 1u);   // round-to-nearest-even
  union { unsigned short s; bf16_t b; } o;
  o.s = (unsigned short)(r >> 16);
  return o.b;
}

DEVI v8f vzero8() {
  v8f c;
#pragma unroll
  for (int i = 0; i < 8; ++i) c[i] = 0.0f;
  return c;
}

DEVI v8f wmma_bf16(v16bf a, v16bf b, v8f c) {
  // D(16x16,f32) = A(16x32,bf16) * B(32x16,bf16) + C
  return __builtin_amdgcn_wmma_f32_16x16x32_bf16(
      /*neg_a=*/false, a, /*neg_b=*/false, b,
      /*c_mod=*/(short)0, c, /*reuse_a=*/false, /*reuse_b=*/false);
}

// A fragment: 16x32 bf16 tile, row-major source with leading dim `ld`.
// lane l: g=l/16, m=l%16; element e -> K = (e<8) ? g*8+e : 16+g*8+(e-8).
DEVI v16bf load_a_bf16(const bf16_t* base, int row0, int col0, int ld) {
  int lane = threadIdx.x & 31;
  int g = lane >> 4, m = lane & 15;
  const bf16_t* p = base + (size_t)(row0 + m) * ld + col0;
  v16bf a;
#pragma unroll
  for (int e = 0; e < 8; ++e) a[e] = p[g * 8 + e];
#pragma unroll
  for (int e = 0; e < 8; ++e) a[8 + e] = p[16 + g * 8 + e];
  return a;
}

// Same A fragment but converting from f32 source on the fly.
DEVI v16bf load_a_f32(const float* base, int row0, int col0, int ld) {
  int lane = threadIdx.x & 31;
  int g = lane >> 4, m = lane & 15;
  const float* p = base + (size_t)(row0 + m) * ld + col0;
  v16bf a;
#pragma unroll
  for (int e = 0; e < 8; ++e) a[e] = f2bf(p[g * 8 + e]);
#pragma unroll
  for (int e = 0; e < 8; ++e) a[8 + e] = f2bf(p[16 + g * 8 + e]);
  return a;
}

// B fragment: 32x16 bf16, source row-major B[k][n] with leading dim ld.
// lane l: g=l/16, n=l%16; element e -> K = (e<8) ? g*8+e : 16+g*8+(e-8). Works on global or LDS ptrs.
DEVI v16bf load_b_rm(const bf16_t* base, int k0, int n0, int ld) {
  int lane = threadIdx.x & 31;
  int g = lane >> 4, n = lane & 15;
  const bf16_t* p = base + (size_t)k0 * ld + n0 + n;
  v16bf b;
#pragma unroll
  for (int e = 0; e < 8; ++e) b[e] = p[(size_t)(g * 8 + e) * ld];
#pragma unroll
  for (int e = 0; e < 8; ++e) b[8 + e] = p[(size_t)(16 + g * 8 + e) * ld];
  return b;
}

DEVI float redmax16(float v) {
  v = fmaxf(v, __shfl_xor(v, 1));
  v = fmaxf(v, __shfl_xor(v, 2));
  v = fmaxf(v, __shfl_xor(v, 4));
  v = fmaxf(v, __shfl_xor(v, 8));
  return v;
}
DEVI float redsum16(float v) {
  v += __shfl_xor(v, 1);
  v += __shfl_xor(v, 2);
  v += __shfl_xor(v, 4);
  v += __shfl_xor(v, 8);
  return v;
}

// ---------------------------------------------------------------------------
// Pack raw Q (head-split) and the reshape-not-transpose Kr, both to bf16.
// Qh[bh][l][d]  = query[b][l][h*64+d]
// Kr[bh][dp][k] = key[b][dp*32 + k/64][h*64 + (k%64)]
// ---------------------------------------------------------------------------
__global__ void pack_qk_kernel(const float* __restrict__ query,
                               const float* __restrict__ key,
                               bf16_t* __restrict__ Qh, bf16_t* __restrict__ Kr) {
  int gid = blockIdx.x * blockDim.x + threadIdx.x;   // 0 .. 2^21-1
  int bh = gid >> 17;
  int b = bh >> 3, h = bh & 7;
  {
    int d = gid & 63;
    int l = (gid >> 6) & 2047;
    Qh[gid] = f2bf(query[((size_t)b * LSEQ + l) * DMODEL + h * DHEAD + d]);
  }
  {
    int k = gid & 2047;
    int dp = (gid >> 11) & 63;
    int lk = dp * 32 + (k >> 6);
    int dd = k & 63;
    Kr[gid] = f2bf(key[((size_t)b * LSEQ + lk) * DMODEL + h * DHEAD + dd]);
  }
}

// Pack W^T in bf16:  T[d*512+o] = W[o*512+d]  for the 4 weight matrices.
__global__ void pack_w_kernel(const float* __restrict__ wq, const float* __restrict__ wk,
                              const float* __restrict__ wv, const float* __restrict__ wo,
                              bf16_t* __restrict__ tq, bf16_t* __restrict__ tk,
                              bf16_t* __restrict__ tv, bf16_t* __restrict__ to) {
  int gid = blockIdx.x * blockDim.x + threadIdx.x;   // 0 .. 4*512*512-1
  int which = gid >> 18;
  int idx = gid & 262143;
  int d = idx >> 9, o = idx & 511;
  const float* W = (which == 0) ? wq : (which == 1) ? wk : (which == 2) ? wv : wo;
  bf16_t* T = (which == 0) ? tq : (which == 1) ? tk : (which == 2) ? tv : to;
  T[idx] = f2bf(W[o * DMODEL + d]);
}

// ---------------------------------------------------------------------------
// Streaming max/mean scoring: M[bh][q] = max_k(QK) - mean_k(QK),
// QK = Qh(2048x64) @ Kr(64x2048) via bf16 WMMA; QK never stored.
// ---------------------------------------------------------------------------
__global__ void mscore_kernel(const bf16_t* __restrict__ Qh,
                              const bf16_t* __restrict__ Kr,
                              float* __restrict__ Msc) {
  int wid = threadIdx.x >> 5, lane = threadIdx.x & 31;
  int wg = blockIdx.x * 8 + wid;          // 0..2047
  int bh = wg >> 7, qtile = wg & 127;
  const bf16_t* Qb = Qh + (size_t)bh * LSEQ * DHEAD;
  const bf16_t* Kb = Kr + (size_t)bh * DHEAD * LSEQ;

  v16bf a0 = load_a_bf16(Qb, qtile * 16, 0, DHEAD);
  v16bf a1 = load_a_bf16(Qb, qtile * 16, 32, DHEAD);

  float lmax[8], lsum[8];
#pragma unroll
  for (int j = 0; j < 8; ++j) { lmax[j] = -3.0e38f; lsum[j] = 0.0f; }

  for (int nt = 0; nt < LSEQ / 16; ++nt) {
    v16bf b0 = load_b_rm(Kb, 0, nt * 16, LSEQ);
    v16bf b1 = load_b_rm(Kb, 32, nt * 16, LSEQ);
    v8f c = vzero8();
    c = wmma_bf16(a0, b0, c);
    c = wmma_bf16(a1, b1, c);
#pragma unroll
    for (int j = 0; j < 8; ++j) { lmax[j] = fmaxf(lmax[j], c[j]); lsum[j] += c[j]; }
  }
#pragma unroll
  for (int j = 0; j < 8; ++j) { lmax[j] = redmax16(lmax[j]); lsum[j] = redsum16(lsum[j]); }

  if ((lane & 15) == 0) {
    int g = lane >> 4;
#pragma unroll
    for (int j = 0; j < 8; ++j) {
      int row = qtile * 16 + g * 8 + j;
      Msc[bh * LSEQ + row] = lmax[j] - lsum[j] * (1.0f / (float)LSEQ);
    }
  }
}

// ---------------------------------------------------------------------------
// Top-7 per (b,h): 7 iterative argmax passes (jax top_k tie-break: low index).
// ---------------------------------------------------------------------------
__global__ void topk_kernel(const float* __restrict__ Msc,
                            int* __restrict__ topidx, int* __restrict__ rowmap) {
  int bh = blockIdx.x;
  int tid = threadIdx.x;
  __shared__ float sv[LSEQ];
  __shared__ float bv[256];
  __shared__ int   bi[256];
  for (int i = tid; i < LSEQ; i += 256) {
    sv[i] = Msc[bh * LSEQ + i];
    rowmap[bh * LSEQ + i] = -1;
  }
  __syncthreads();
  for (int t = 0; t < NTOP; ++t) {
    float best = -3.0e38f; int bidx = 1 << 30;
    for (int i = tid; i < LSEQ; i += 256) {
      float v = sv[i];
      if (v > best || (v == best && i < bidx)) { best = v; bidx = i; }
    }
    bv[tid] = best; bi[tid] = bidx;
    __syncthreads();
    if (tid == 0) {
      float b = bv[0]; int x = bi[0];
      for (int i = 1; i < 256; ++i)
        if (bv[i] > b || (bv[i] == b && bi[i] < x)) { b = bv[i]; x = bi[i]; }
      topidx[bh * NTOP + t] = x;
      sv[x] = -3.0e38f;
      rowmap[bh * LSEQ + x] = t;
    }
    __syncthreads();
  }
}

// ---------------------------------------------------------------------------
// QK_red rows (7 per head) in full f32 straight from the raw inputs.
// ---------------------------------------------------------------------------
__global__ void maskrows_kernel(const float* __restrict__ query,
                                const float* __restrict__ key,
                                const int* __restrict__ topidx,
                                float* __restrict__ maskr) {
  int gid = blockIdx.x * blockDim.x + threadIdx.x;  // 16*7*2048 total
  if (gid >= BH * NTOP * LSEQ) return;
  int k = gid & 2047;
  int t = (gid >> 11) % NTOP;
  int bh = gid / (NTOP * LSEQ);
  int b = bh >> 3, h = bh & 7;
  int qrow = topidx[bh * NTOP + t];
  const float* qv = query + ((size_t)b * LSEQ + qrow) * DMODEL + h * DHEAD;
  float acc = 0.0f;
#pragma unroll 4
  for (int dp = 0; dp < DHEAD; ++dp) {
    int lk = dp * 32 + (k >> 6);
    int dd = k & 63;
    acc += qv[dp] * key[((size_t)b * LSEQ + lk) * DMODEL + h * DHEAD + dd];
  }
  maskr[gid] = acc;
}

// ---------------------------------------------------------------------------
// Projection GEMM: OUT[bh][l][d] = bf16( X(4096x512,f32) @ W^T + bias ),
// scattered into head-split [BH][LSEQ][DHEAD] layout. One wave per 16x16 tile.
// ---------------------------------------------------------------------------
__global__ void proj_kernel(const float* __restrict__ X, const bf16_t* __restrict__ WT,
                            const float* __restrict__ bias, bf16_t* __restrict__ OUT) {
  int wid = threadIdx.x >> 5, lane = threadIdx.x & 31;
  int wg = blockIdx.x * 8 + wid;           // 0..8191
  int mtile = wg >> 5, ntile = wg & 31;

  v8f c = vzero8();
  for (int ks = 0; ks < DMODEL / 32; ++ks) {
    int ksn = (ks + 1 < DMODEL / 32) ? ks + 1 : ks;
    __builtin_prefetch(WT + (size_t)(ksn * 32 + lane) * DMODEL + ntile * 16, 0, 0);
    v16bf a = load_a_f32(X, mtile * 16, ks * 32, DMODEL);
    v16bf b = load_b_rm(WT, ks * 32, ntile * 16, DMODEL);
    c = wmma_bf16(a, b, c);
  }
  int g = lane >> 4, m = lane & 15;
  int n = ntile * 16 + m;
  int h = n >> 6, dcol = n & 63;
  float bb = bias[n];
#pragma unroll
  for (int j = 0; j < 8; ++j) {
    int r = mtile * 16 + g * 8 + j;
    int b_ = r >> 11, lq = r & 2047;
    OUT[(((size_t)(b_ * NHEAD + h)) * LSEQ + lq) * DHEAD + dcol] = f2bf(c[j] + bb);
  }
}

// ---------------------------------------------------------------------------
// Flash-style attention with ProbSparse additive mask.
// Block = 8 waves = 8 consecutive 16-query tiles of the SAME (b,h).
// Per 32-key step: K tile staged coalesced->LDS transposed ([64 d][32 key]),
// V tile staged via async global->LDS (double buffered, ASYNCcnt), then
// S (4 wmma from LDS) -> online softmax -> P via per-wave LDS -> PV (4 wmma).
// ---------------------------------------------------------------------------
__global__ void attn_kernel(const bf16_t* __restrict__ qp, const bf16_t* __restrict__ kpj,
                            const bf16_t* __restrict__ vp,
                            const int* __restrict__ rowmap, const float* __restrict__ maskr,
                            bf16_t* __restrict__ att) {
  __shared__ bf16_t KtT[2][DHEAD * 32];   // [d][key] transposed, 4KB each
  __shared__ bf16_t Vt [2][32 * DHEAD];   // [key][d] natural,    4KB each
  __shared__ bf16_t Pbuf[8][16 * 32];     // per-wave P tiles,    8KB

  int tid = threadIdx.x;
  int wid = tid >> 5, lane = tid & 31;
  int g = lane >> 4, m = lane & 15;
  int wg = blockIdx.x * 8 + wid;          // 0..2047
  int bh = wg >> 7, qtile = wg & 127;     // all waves in block share bh
  int b = bh >> 3, h = bh & 7;

  const bf16_t* Qb = qp  + (size_t)bh * LSEQ * DHEAD;
  const bf16_t* Kb = kpj + (size_t)bh * LSEQ * DHEAD;
  const bf16_t* Vb = vp  + (size_t)bh * LSEQ * DHEAD;

  // cooperative staging coordinates: thread -> (key row, 8-wide d chunk)
  int skey = tid >> 3;                    // 0..31
  int sdc  = (tid & 7) * 8;               // 0,8,...,56

  v16bf aq0 = load_a_bf16(Qb, qtile * 16, 0, DHEAD);
  v16bf aq1 = load_a_bf16(Qb, qtile * 16, 32, DHEAD);

  int slot[8];
#pragma unroll
  for (int j = 0; j < 8; ++j)
    slot[j] = rowmap[bh * LSEQ + qtile * 16 + g * 8 + j];

  float mrow[8], srow[8];
  v8f o[4];
#pragma unroll
  for (int j = 0; j < 8; ++j) { mrow[j] = -3.0e38f; srow[j] = 0.0f; }
#pragma unroll
  for (int nt = 0; nt < 4; ++nt) o[nt] = vzero8();

  const float scale = 0.125f;             // dh^-0.5 = 1/8

  // prologue: stage key-tile 0 into buffer 0
  {
    uint4 kv = *(const uint4*)(Kb + (size_t)skey * DHEAD + sdc);
    const bf16_t* kvh = (const bf16_t*)&kv;
#pragma unroll
    for (int e = 0; e < 8; ++e) KtT[0][(sdc + e) * 32 + skey] = kvh[e];
    async_cp16(&Vt[0][skey * DHEAD + sdc], Vb + (size_t)skey * DHEAD + sdc);
  }

  for (int it = 0; it < LSEQ / 32; ++it) {
    int cur = it & 1;
    int kt = it * 32;
    async_wait_all();                     // V tile of this iter landed in LDS
    __syncthreads();                      // staging visible to all waves

    if (it + 1 < LSEQ / 32) {             // stage next tile into other buffer
      int nb = cur ^ 1, nkt = kt + 32;
      uint4 kv = *(const uint4*)(Kb + (size_t)(nkt + skey) * DHEAD + sdc);
      const bf16_t* kvh = (const bf16_t*)&kv;
#pragma unroll
      for (int e = 0; e < 8; ++e) KtT[nb][(sdc + e) * 32 + skey] = kvh[e];
      async_cp16(&Vt[nb][skey * DHEAD + sdc], Vb + (size_t)(nkt + skey) * DHEAD + sdc);
    }

    const bf16_t* Kl = KtT[cur];
    const bf16_t* Vl = Vt[cur];

    // ---- S tiles: [16q x 32k], B fragments read from [d][key] LDS at stride 32 ----
    v8f c0 = vzero8(), c1 = vzero8();
    c0 = wmma_bf16(aq0, load_b_rm(Kl, 0,  0,  32), c0);
    c0 = wmma_bf16(aq1, load_b_rm(Kl, 32, 0,  32), c0);
    c1 = wmma_bf16(aq0, load_b_rm(Kl, 0,  16, 32), c1);
    c1 = wmma_bf16(aq1, load_b_rm(Kl, 32, 16, 32), c1);

    // ---- scale + ProbSparse mask + online softmax ----
    float p0[8], p1[8], resc[8];
#pragma unroll
    for (int j = 0; j < 8; ++j) {
      float v0 = c0[j] * scale, v1 = c1[j] * scale;
      if (slot[j] >= 0) {
        const float* mr = maskr + ((size_t)bh * NTOP + slot[j]) * LSEQ + kt + m;
        v0 += mr[0];
        v1 += mr[16];
      }
      float tmax = redmax16(fmaxf(v0, v1));
      float nmax = fmaxf(mrow[j], tmax);
      resc[j] = __expf(mrow[j] - nmax);
      p0[j] = __expf(v0 - nmax);
      p1[j] = __expf(v1 - nmax);
      srow[j] = srow[j] * resc[j] + redsum16(p0[j] + p1[j]);
      mrow[j] = nmax;
    }
#pragma unroll
    for (int nt = 0; nt < 4; ++nt)
#pragma unroll
      for (int j = 0; j < 8; ++j) o[nt][j] *= resc[j];

    // ---- C-layout P -> A-layout fragment via per-wave LDS tile ----
    bf16_t* Pl = Pbuf[wid];
#pragma unroll
    for (int j = 0; j < 8; ++j) {
      Pl[(g * 8 + j) * 32 + m]      = f2bf(p0[j]);
      Pl[(g * 8 + j) * 32 + 16 + m] = f2bf(p1[j]);
    }
    __syncthreads();
    v16bf ap;
#pragma unroll
    for (int e = 0; e < 8; ++e) ap[e]     = Pl[m * 32 + g * 8 + e];
#pragma unroll
    for (int e = 0; e < 8; ++e) ap[8 + e] = Pl[m * 32 + 16 + g * 8 + e];
    __syncthreads();

    // ---- PV: P(16x32) @ V(32x64) from LDS, four 16-wide d tiles ----
#pragma unroll
    for (int nt = 0; nt < 4; ++nt) {
      v16bf bv = load_b_rm(Vl, 0, nt * 16, DHEAD);
      o[nt] = wmma_bf16(ap, bv, o[nt]);
    }
  }

  // ---- normalize and store to [ROWS x DMODEL] bf16 activation buffer ----
  float inv[8];
#pragma unroll
  for (int j = 0; j < 8; ++j) inv[j] = 1.0f / srow[j];
#pragma unroll
  for (int nt = 0; nt < 4; ++nt) {
    int col = h * DHEAD + nt * 16 + m;
#pragma unroll
    for (int j = 0; j < 8; ++j) {
      int r = b * LSEQ + qtile * 16 + g * 8 + j;
      att[(size_t)r * DMODEL + col] = f2bf(o[nt][j] * inv[j]);
    }
  }
}

// ---------------------------------------------------------------------------
// Output projection: d_out(f32) = att(bf16) @ wo^T + bo
// ---------------------------------------------------------------------------
__global__ void outproj_kernel(const bf16_t* __restrict__ att, const bf16_t* __restrict__ WT,
                               const float* __restrict__ bias, float* __restrict__ out) {
  int wid = threadIdx.x >> 5, lane = threadIdx.x & 31;
  int wg = blockIdx.x * 8 + wid;
  int mtile = wg >> 5, ntile = wg & 31;

  v8f c = vzero8();
  for (int ks = 0; ks < DMODEL / 32; ++ks) {
    int ksn = (ks + 1 < DMODEL / 32) ? ks + 1 : ks;
    __builtin_prefetch(WT + (size_t)(ksn * 32 + lane) * DMODEL + ntile * 16, 0, 0);
    v16bf a = load_a_bf16(att, mtile * 16, ks * 32, DMODEL);
    v16bf b = load_b_rm(WT, ks * 32, ntile * 16, DMODEL);
    c = wmma_bf16(a, b, c);
  }
  int g = lane >> 4, m = lane & 15;
  int n = ntile * 16 + m;
  float bb = bias[n];
#pragma unroll
  for (int j = 0; j < 8; ++j) {
    int r = mtile * 16 + g * 8 + j;
    out[(size_t)r * DMODEL + n] = c[j] + bb;
  }
}

// ---------------------------------------------------------------------------
extern "C" void kernel_launch(void* const* d_in, const int* in_sizes, int n_in,
                              void* d_out, int out_size, void* d_ws, size_t ws_size,
                              hipStream_t stream) {
  (void)in_sizes; (void)n_in; (void)out_size; (void)ws_size;
  const float* query = (const float*)d_in[0];
  const float* key   = (const float*)d_in[1];
  const float* value = (const float*)d_in[2];
  const float* wq    = (const float*)d_in[3];
  const float* wk    = (const float*)d_in[4];
  const float* wv    = (const float*)d_in[5];
  const float* bq    = (const float*)d_in[6];
  const float* bk    = (const float*)d_in[7];
  const float* bv    = (const float*)d_in[8];
  const float* wo    = (const float*)d_in[9];
  const float* bo    = (const float*)d_in[10];

  char* ws = (char*)d_ws;
  size_t off = 0;
  auto take = [&](size_t bytes) -> void* {
    void* p = ws + off;
    off += (bytes + 255) & ~(size_t)255;
    return p;
  };
  bf16_t* Qh   = (bf16_t*)take((size_t)BH * LSEQ * DHEAD * 2);
  bf16_t* Kr   = (bf16_t*)take((size_t)BH * DHEAD * LSEQ * 2);
  bf16_t* qp   = (bf16_t*)take((size_t)BH * LSEQ * DHEAD * 2);
  bf16_t* kpj  = (bf16_t*)take((size_t)BH * LSEQ * DHEAD * 2);
  bf16_t* vp   = (bf16_t*)take((size_t)BH * LSEQ * DHEAD * 2);
  bf16_t* wqT  = (bf16_t*)take((size_t)DMODEL * DMODEL * 2);
  bf16_t* wkT  = (bf16_t*)take((size_t)DMODEL * DMODEL * 2);
  bf16_t* wvT  = (bf16_t*)take((size_t)DMODEL * DMODEL * 2);
  bf16_t* woT  = (bf16_t*)take((size_t)DMODEL * DMODEL * 2);
  float*  Msc  = (float*)take((size_t)BH * LSEQ * 4);
  int*    topi = (int*)take((size_t)BH * NTOP * 4);
  int*    rmap = (int*)take((size_t)BH * LSEQ * 4);
  float*  mskr = (float*)take((size_t)BH * NTOP * LSEQ * 4);
  bf16_t* att  = (bf16_t*)take((size_t)ROWS * DMODEL * 2);

  pack_qk_kernel<<<8192, 256, 0, stream>>>(query, key, Qh, Kr);
  pack_w_kernel<<<4096, 256, 0, stream>>>(wq, wk, wv, wo, wqT, wkT, wvT, woT);
  mscore_kernel<<<256, 256, 0, stream>>>(Qh, Kr, Msc);
  topk_kernel<<<BH, 256, 0, stream>>>(Msc, topi, rmap);
  maskrows_kernel<<<(BH * NTOP * LSEQ + 255) / 256, 256, 0, stream>>>(query, key, topi, mskr);
  proj_kernel<<<1024, 256, 0, stream>>>(query, wqT, bq, qp);
  proj_kernel<<<1024, 256, 0, stream>>>(key,   wkT, bk, kpj);
  proj_kernel<<<1024, 256, 0, stream>>>(value, wvT, bv, vp);
  attn_kernel<<<256, 256, 0, stream>>>(qp, kpj, vp, rmap, mskr, att);
  outproj_kernel<<<1024, 256, 0, stream>>>(att, woT, bo, (float*)d_out);
}
